// LearnerEncoder_18811956757093
// MI455X (gfx1250) — matrix operand order repaired and checked by reading, not compiled, not run on hardware
//
#include <hip/hip_runtime.h>

typedef __attribute__((ext_vector_type(16))) __bf16 v16bf;
typedef __attribute__((ext_vector_type(8)))  float  v8f;
typedef __attribute__((ext_vector_type(4)))  int    v4i;

#define D_MODEL 512
#define DINNER  1024
#define NHEAD   8
#define HDIM    64

// ---------------------------------------------------------------------------
// gfx1250 async global->LDS copy (ASYNCcnt path), fallback: plain vec copy.
// ---------------------------------------------------------------------------
#if defined(__has_builtin)
#if __has_builtin(__builtin_amdgcn_global_load_async_to_lds_b128)
#define HAVE_ASYNC_LDS 1
#endif
#endif
#ifndef HAVE_ASYNC_LDS
#define HAVE_ASYNC_LDS 0
#endif

typedef __attribute__((address_space(1))) v4i* gptr4;
typedef __attribute__((address_space(3))) v4i* lptr4;

__device__ __forceinline__ void async_cp16(void* lds_dst, const void* gsrc) {
#if HAVE_ASYNC_LDS
  __builtin_amdgcn_global_load_async_to_lds_b128((gptr4)gsrc, (lptr4)lds_dst,
                                                 0, 0);
#else
  *(int4*)lds_dst = *(const int4*)gsrc;
#endif
}

__device__ __forceinline__ void async_wait() {
#if HAVE_ASYNC_LDS
#if __has_builtin(__builtin_amdgcn_s_wait_asynccnt)
  __builtin_amdgcn_s_wait_asynccnt(0);
#else
  asm volatile("s_wait_asynccnt 0x0" ::: "memory");
#endif
#endif
}

// ---------------------------------------------------------------------------
// WMMA fragment helpers (CDNA5 16x16x32 bf16 layouts, ISA 7.12.2)
// A (16x32): lane<16: K0..7 (e0-7), K16..23 (e8-15); lane>=16: K8..15, K24..31
// B (32x16): lane<16 col n: K0..7 (e0-7), K16..23 (e8-15); lane>=16: K8..15, K24..31
// C/D 16x16 f32: lanes 0-15: M=v, N=lane; lanes 16-31: M=8+v, N=lane-16
// ---------------------------------------------------------------------------
__device__ __forceinline__ v16bf frag_a(const __bf16* base, int ld, int row,
                                        int kBase, int lane) {
  const __bf16* p = base + (size_t)row * ld + kBase + ((lane >> 4) << 3);
  v16bf f;
#pragma unroll
  for (int e = 0; e < 8; ++e) { f[e] = p[e]; f[e + 8] = p[e + 16]; }
  return f;
}

__device__ __forceinline__ v16bf frag_b(const __bf16* base, int ld, int n,
                                        int kBase, int lane) {
  const __bf16* p = base + (size_t)n * ld + kBase + ((lane >> 4) << 3);
  v16bf f;
#pragma unroll
  for (int e = 0; e < 8; ++e) { f[e] = p[e]; f[e + 8] = p[e + 16]; }
  return f;
}

__device__ __forceinline__ v8f wmma_bf16(v16bf a, v16bf b, v8f c) {
  return __builtin_amdgcn_wmma_f32_16x16x32_bf16(false, a, false, b,
                                                 (short)0, c, false, false);
}

// ---------------------------------------------------------------------------
// f32 -> bf16 conversion (plain)
// ---------------------------------------------------------------------------
__global__ void f32_to_bf16_kernel(const float* __restrict__ in,
                                   __bf16* __restrict__ out, int n) {
  for (int i = blockIdx.x * blockDim.x + threadIdx.x; i < n;
       i += gridDim.x * blockDim.x)
    out[i] = (__bf16)in[i];
}

// ---------------------------------------------------------------------------
// f32[K][N] -> bf16[N][K] transposed conversion (weights stored as B^T so
// GEMM B-tile staging is contiguous 16B async chunks).
// ---------------------------------------------------------------------------
__global__ __launch_bounds__(256) void f32_to_bf16_t_kernel(
    const float* __restrict__ in, __bf16* __restrict__ out, int K, int N) {
  __shared__ float t[32][33];
  int bx = blockIdx.x * 32;  // n
  int by = blockIdx.y * 32;  // k
  int tx = threadIdx.x & 31, ty = threadIdx.x >> 5;  // 32x8
#pragma unroll
  for (int i = ty; i < 32; i += 8) t[i][tx] = in[(size_t)(by + i) * N + bx + tx];
  __syncthreads();
#pragma unroll
  for (int i = ty; i < 32; i += 8)
    out[(size_t)(bx + i) * K + by + tx] = (__bf16)t[tx][i];
}

// ---------------------------------------------------------------------------
// Embedding gather + positional add -> bf16 tokens (pos idx = t % seqLen)
// ---------------------------------------------------------------------------
__global__ __launch_bounds__(256) void embed_kernel(
    const int* __restrict__ ids, const float* __restrict__ emb,
    const float* __restrict__ pos, __bf16* __restrict__ out, int seqLen) {
  int t = blockIdx.x;
  int id = ids[t];
  const float* e = emb + (size_t)id * D_MODEL;
  const float* p = pos + (size_t)(t % seqLen) * D_MODEL;
  __bf16* o = out + (size_t)t * D_MODEL;
  for (int c = threadIdx.x; c < D_MODEL; c += blockDim.x)
    o[c] = (__bf16)(e[c] + p[c]);
}

// ---------------------------------------------------------------------------
// Tiled GEMM: C[M,N] = A[M,K] x B[K,N], B given transposed as Bt[N][K].
// 64x64x64 tile, 256 threads (8 wave32), double-buffered async global->LDS
// staging so tile t+1 loads overlap tile t WMMAs.  flags bit0: relu.
// GUARD=true adds M-edge handling (only needed for the M=16 GEMMs).
// ---------------------------------------------------------------------------
template <bool GUARD>
__global__ __launch_bounds__(256) void gemm_bf16_kernel(
    const __bf16* __restrict__ A, const __bf16* __restrict__ Bt,
    __bf16* __restrict__ C, int M, int N, int K,
    const float* __restrict__ bias, const __bf16* __restrict__ resid,
    int flags) {
  __shared__ __align__(16) __bf16 As[2][64][64];  // 2 x 8 KB
  __shared__ __align__(16) __bf16 Bs[2][64][64];  // 2 x 8 KB ([n][k])

  const int tid = threadIdx.x;
  const int lane = tid & 31;
  const int wave = tid >> 5;
  const int bm = blockIdx.x * 64;
  const int bn = blockIdx.y * 64;
  const int mw = (wave & 3) * 16;   // row strip
  const int nw = (wave >> 2) * 32;  // col base, tiles at +0,+16

  const int sr = tid >> 2;          // 0..63 staging row
  const int sc = (tid & 3) * 16;    // 0,16,32,48 staging col

  auto stage = [&](int buf, int k0) {
    if (!GUARD || (bm + sr) < M) {
      const __bf16* srcA = A + (size_t)(bm + sr) * K + k0 + sc;
      async_cp16(&As[buf][sr][sc], srcA);
      async_cp16(&As[buf][sr][sc + 8], srcA + 8);
    } else {
#pragma unroll
      for (int e = 0; e < 16; ++e) As[buf][sr][sc + e] = (__bf16)0.0f;
    }
    const __bf16* srcB = Bt + (size_t)(bn + sr) * K + k0 + sc;
    async_cp16(&Bs[buf][sr][sc], srcB);
    async_cp16(&Bs[buf][sr][sc + 8], srcB + 8);
  };

  v8f acc[2] = {v8f{}, v8f{}};

  stage(0, 0);
  int cur = 0;
  for (int k0 = 0; k0 < K; k0 += 64) {
    async_wait();      // loads for tile 'cur' complete
    __syncthreads();
    if (k0 + 64 < K) stage(cur ^ 1, k0 + 64);  // overlap with WMMAs below

#pragma unroll
    for (int kk = 0; kk < 64; kk += 32) {
      v16bf a = frag_a(&As[cur][0][0], 64, mw + (lane & 15), kk, lane);
#pragma unroll
      for (int j = 0; j < 2; ++j) {
        v16bf b = frag_b(&Bs[cur][0][0], 64, nw + j * 16 + (lane & 15), kk, lane);
        acc[j] = wmma_bf16(a, b, acc[j]);
      }
    }
    __syncthreads();
    cur ^= 1;
  }

  const int col = lane & 15;
  const int rb = (lane >> 4) * 8;
#pragma unroll
  for (int j = 0; j < 2; ++j) {
    int cn = bn + nw + j * 16 + col;
    float bv = bias ? bias[cn] : 0.0f;
#pragma unroll
    for (int v = 0; v < 8; ++v) {
      int gr = bm + mw + rb + v;
      if (!GUARD || gr < M) {
        float x = acc[j][v] + bv;
        if (resid) x += (float)resid[(size_t)gr * N + cn];
        if (flags & 1) x = fmaxf(x, 0.0f);
        C[(size_t)gr * N + cn] = (__bf16)x;
      }
    }
  }
}

// ---------------------------------------------------------------------------
// Fused attention per (group, head): S = QK^T*scale -> softmax -> O = P V
// Q: [G][64][512], K/V: [G/kvDiv][128][512]; head slice h*64. 90 KB LDS.
// ---------------------------------------------------------------------------
__global__ __launch_bounds__(256) void attn_kernel(
    const __bf16* __restrict__ Q, const __bf16* __restrict__ Kk,
    const __bf16* __restrict__ V, __bf16* __restrict__ O, int kvDiv,
    float scale) {
  extern __shared__ char smem[];
  __bf16* Qs = (__bf16*)smem;          // 64x64   (8 KB)
  __bf16* Ks = Qs + 64 * 64;           // 128x64  (16 KB)
  __bf16* Vt = Ks + 128 * 64;          // 64x128  (16 KB, [dv][key])
  float* S = (float*)(Vt + 64 * 128);  // 64x128 f32 (32 KB)
  __bf16* P = (__bf16*)(S + 64 * 128); // 64x128 (16 KB)

  const int g = blockIdx.x, h = blockIdx.y;
  const int gk = g / kvDiv;
  const __bf16* Qg = Q + ((size_t)g * 64) * D_MODEL + h * HDIM;
  const __bf16* Kg = Kk + ((size_t)gk * 128) * D_MODEL + h * HDIM;
  const __bf16* Vg = V + ((size_t)gk * 128) * D_MODEL + h * HDIM;
  const int tid = threadIdx.x, lane = tid & 31, wave = tid >> 5;

  {  // stage Q 64x64 (async, contiguous 16B chunks)
    int r = tid >> 2, c = (tid & 3) * 16;
    const __bf16* src = Qg + (size_t)r * D_MODEL + c;
    async_cp16(Qs + r * 64 + c, src);
    async_cp16(Qs + r * 64 + c + 8, src + 8);
  }
  {  // stage K 128x64 (async)
    int r = tid >> 1, c = (tid & 1) * 32;
    const __bf16* src = Kg + (size_t)r * D_MODEL + c;
    async_cp16(Ks + r * 64 + c, src);
    async_cp16(Ks + r * 64 + c + 8, src + 8);
    async_cp16(Ks + r * 64 + c + 16, src + 16);
    async_cp16(Ks + r * 64 + c + 24, src + 24);
  }
  {  // stage V transposed -> Vt[dv][key] (one-time LDS scatter)
    int r = tid >> 1, c = (tid & 1) * 32;
#pragma unroll
    for (int e = 0; e < 32; ++e) Vt[(c + e) * 128 + r] = Vg[(size_t)r * D_MODEL + c + e];
  }
  async_wait();
  __syncthreads();

  {  // S = Q K^T : M=64, N=128(keys), K=64
    const int mq = (wave & 3) * 16, nb = (wave >> 2) * 64;
    v8f acc[4] = {v8f{}, v8f{}, v8f{}, v8f{}};
#pragma unroll
    for (int k0 = 0; k0 < 64; k0 += 32) {
      v16bf a = frag_a(Qs, 64, mq + (lane & 15), k0, lane);
#pragma unroll
      for (int j = 0; j < 4; ++j) {
        v16bf b = frag_b(Ks, 64, nb + j * 16 + (lane & 15), k0, lane);
        acc[j] = wmma_bf16(a, b, acc[j]);
      }
    }
    const int col = lane & 15, rb = (lane >> 4) * 8;
#pragma unroll
    for (int j = 0; j < 4; ++j)
#pragma unroll
      for (int v = 0; v < 8; ++v)
        S[(mq + rb + v) * 128 + nb + j * 16 + col] = acc[j][v] * scale;
  }
  __syncthreads();

  if (tid < 64) {  // row softmax
    float mx = -1e30f;
    for (int i = 0; i < 128; ++i) mx = fmaxf(mx, S[tid * 128 + i]);
    float sum = 0.0f;
    for (int i = 0; i < 128; ++i) {
      float e = __expf(S[tid * 128 + i] - mx);
      S[tid * 128 + i] = e;
      sum += e;
    }
    float inv = 1.0f / sum;
    for (int i = 0; i < 128; ++i) P[tid * 128 + i] = (__bf16)(S[tid * 128 + i] * inv);
  }
  __syncthreads();

  {  // O = P V : M=64, N=64(dv), K=128(keys)
    const int mq = (wave & 3) * 16, nb = (wave >> 2) * 32;
    v8f acc[2] = {v8f{}, v8f{}};
#pragma unroll
    for (int k0 = 0; k0 < 128; k0 += 32) {
      v16bf a = frag_a(P, 128, mq + (lane & 15), k0, lane);
#pragma unroll
      for (int j = 0; j < 2; ++j) {
        v16bf b = frag_b(Vt, 128, nb + j * 16 + (lane & 15), k0, lane);
        acc[j] = wmma_bf16(a, b, acc[j]);
      }
    }
    const int col = lane & 15, rb = (lane >> 4) * 8;
    __bf16* Og = O + ((size_t)g * 64) * D_MODEL + h * HDIM;
#pragma unroll
    for (int j = 0; j < 2; ++j)
#pragma unroll
      for (int v = 0; v < 8; ++v)
        Og[(size_t)(mq + rb + v) * D_MODEL + nb + j * 16 + col] = (__bf16)acc[j][v];
  }
}

// ---------------------------------------------------------------------------
// LayerNorm over 512 cols; one block per token; bf16 in, bf16 or f32 out.
// ---------------------------------------------------------------------------
__global__ __launch_bounds__(256) void ln_kernel(
    const __bf16* __restrict__ X, const float* __restrict__ gam,
    const float* __restrict__ bet, void* __restrict__ out, int out_f32) {
  __shared__ float red[256];
  const int t = blockIdx.x, tid = threadIdx.x;
  const __bf16* x = X + (size_t)t * D_MODEL;
  float v0 = (float)x[tid], v1 = (float)x[tid + 256];
  red[tid] = v0 + v1;
  __syncthreads();
  for (int s = 128; s > 0; s >>= 1) {
    if (tid < s) red[tid] += red[tid + s];
    __syncthreads();
  }
  float mean = red[0] * (1.0f / 512.0f);
  __syncthreads();
  float d0 = v0 - mean, d1 = v1 - mean;
  red[tid] = d0 * d0 + d1 * d1;
  __syncthreads();
  for (int s = 128; s > 0; s >>= 1) {
    if (tid < s) red[tid] += red[tid + s];
    __syncthreads();
  }
  float inv = rsqrtf(red[0] * (1.0f / 512.0f) + 1e-6f);
  float o0 = d0 * inv * gam[tid] + bet[tid];
  float o1 = d1 * inv * gam[tid + 256] + bet[tid + 256];
  if (out_f32) {
    float* o = (float*)out + (size_t)t * D_MODEL;
    o[tid] = o0;
    o[tid + 256] = o1;
  } else {
    __bf16* o = (__bf16*)out + (size_t)t * D_MODEL;
    o[tid] = (__bf16)o0;
    o[tid + 256] = (__bf16)o1;
  }
}

// ---------------------------------------------------------------------------
// Mcat[16][31][128][512]: rows 0..63 <- M[b][0], rows 64..127 <- M[b][i+1]
// ---------------------------------------------------------------------------
__global__ __launch_bounds__(256) void mcat_kernel(const __bf16* __restrict__ M,
                                                   __bf16* __restrict__ Mc) {
  int idx = blockIdx.x;  // 16*31*128 rows
  int r = idx & 127;
  int bi = idx >> 7;
  int i = bi % 31, b = bi / 31;
  int srcObj = (r < 64) ? 0 : (i + 1);
  const __bf16* src = M + (((size_t)(b * 32 + srcObj) * 64 + (r & 63))) * D_MODEL;
  __bf16* dst = Mc + (size_t)idx * D_MODEL;
  for (int c = threadIdx.x; c < D_MODEL; c += blockDim.x) dst[c] = src[c];
}

// ---------------------------------------------------------------------------
// Block3 attention collapse: X[t] = I[t] + r[t / (31*64)]
// ---------------------------------------------------------------------------
__global__ __launch_bounds__(256) void bcast_add_kernel(
    const __bf16* __restrict__ I, const __bf16* __restrict__ r,
    __bf16* __restrict__ X) {
  int t = blockIdx.x;
  int b = t / (31 * 64);
  const __bf16* ip = I + (size_t)t * D_MODEL;
  const __bf16* rp = r + (size_t)b * D_MODEL;
  __bf16* xp = X + (size_t)t * D_MODEL;
  for (int c = threadIdx.x; c < D_MODEL; c += blockDim.x)
    xp[c] = (__bf16)((float)ip[c] + (float)rp[c]);
}

// ---------------------------------------------------------------------------
// Host orchestration
// ---------------------------------------------------------------------------
extern "C" void kernel_launch(void* const* d_in, const int* in_sizes, int n_in,
                              void* d_out, int out_size, void* d_ws,
                              size_t ws_size, hipStream_t stream) {
  (void)in_sizes; (void)n_in; (void)out_size; (void)ws_size;

  const int BS = 16, NOBJ = 32, LM = 128, LOBJ = 64, LINT = 64;
  const int NT_M = BS * LM;                      // 2048
  const int NT_O = BS * NOBJ * LOBJ;             // 32768
  const int NT_I = BS * (NOBJ - 1) * LINT;       // 31744
  const int NT_MC = BS * (NOBJ - 1) * 2 * LOBJ;  // 63488

  const float* map_embeds = (const float*)d_in[0];
  const int* motion_ids = (const int*)d_in[1];
  const int* object_ids = (const int*)d_in[2];
  const int* interaction_ids = (const int*)d_in[3];
  const float* emb = (const float*)d_in[4];
  const float* pos = (const float*)d_in[5];

  // --- workspace arena ---
  char* ws = (char*)d_ws;
  const size_t WBLK_ELEMS = 4 * (512 * 512) + 2 * (512 * 1024);  // 2,097,152
  __bf16* wb = (__bf16*)ws;
  const size_t WBYTES = 3 * WBLK_ELEMS * sizeof(__bf16);  // 12 MB
  const size_t SLOT = 64ull * 1024 * 1024;
  char* A0 = ws + WBYTES;
  char* A1 = A0 + SLOT;
  char* A2 = A1 + SLOT;
  char* A3 = A2 + SLOT;
  char* A4 = A3 + SLOT;
  char* A5 = A4 + SLOT;
  char* A6 = A5 + SLOT;

  struct BW {
    __bf16 *wq, *wk, *wv, *wo, *w1, *w2;  // all stored transposed [N][K]
    const float *ln1g, *ln1b, *b1, *b2, *ln2g, *ln2b;
  } blk[3];
  auto cvtT = [&](const float* s, __bf16* d, int K, int N) {
    dim3 g(N / 32, K / 32);
    f32_to_bf16_t_kernel<<<g, 256, 0, stream>>>(s, d, K, N);
  };
  for (int b = 0; b < 3; ++b) {
    __bf16* base = wb + (size_t)b * WBLK_ELEMS;
    blk[b].wq = base;
    blk[b].wk = base + 262144;
    blk[b].wv = base + 524288;
    blk[b].wo = base + 786432;
    blk[b].w1 = base + 1048576;
    blk[b].w2 = base + 1572864;
    int ib = 6 + b * 12;
    blk[b].ln1g = (const float*)d_in[ib + 4];
    blk[b].ln1b = (const float*)d_in[ib + 5];
    blk[b].b1 = (const float*)d_in[ib + 7];
    blk[b].b2 = (const float*)d_in[ib + 9];
    blk[b].ln2g = (const float*)d_in[ib + 10];
    blk[b].ln2b = (const float*)d_in[ib + 11];
    cvtT((const float*)d_in[ib + 0], blk[b].wq, 512, 512);
    cvtT((const float*)d_in[ib + 1], blk[b].wk, 512, 512);
    cvtT((const float*)d_in[ib + 2], blk[b].wv, 512, 512);
    cvtT((const float*)d_in[ib + 3], blk[b].wo, 512, 512);
    cvtT((const float*)d_in[ib + 6], blk[b].w1, 512, 1024);
    cvtT((const float*)d_in[ib + 8], blk[b].w2, 1024, 512);
  }

  auto gemm = [&](const __bf16* A, const __bf16* Bt, __bf16* C, int M, int N,
                  int K, const float* bias, const __bf16* resid, int flags) {
    dim3 g((M + 63) / 64, N / 64);
    if ((M & 63) == 0)
      gemm_bf16_kernel<false><<<g, 256, 0, stream>>>(A, Bt, C, M, N, K, bias,
                                                     resid, flags);
    else
      gemm_bf16_kernel<true><<<g, 256, 0, stream>>>(A, Bt, C, M, N, K, bias,
                                                    resid, flags);
  };
  const unsigned ATTN_SMEM =
      (64 * 64 + 128 * 64 + 64 * 128) * 2 + 64 * 128 * 4 + 64 * 128 * 2;  // 90112

  // ================= Block 1 (p_motion) =================
  __bf16* Xo = (__bf16*)A0;
  __bf16* Xm = (__bf16*)A1;
  __bf16* Km = (__bf16*)A2;
  __bf16* Vm = (__bf16*)A3;
  __bf16* Qo = (__bf16*)A4;
  __bf16* O1 = (__bf16*)A5;
  embed_kernel<<<NT_M, 256, 0, stream>>>(motion_ids, emb, pos, Xm, LM);
  embed_kernel<<<NT_O, 256, 0, stream>>>(object_ids, emb, pos, Xo, LOBJ);
  gemm(Xm, blk[0].wk, Km, NT_M, 512, 512, nullptr, nullptr, 0);
  gemm(Xm, blk[0].wv, Vm, NT_M, 512, 512, nullptr, nullptr, 0);
  gemm(Xo, blk[0].wq, Qo, NT_O, 512, 512, nullptr, nullptr, 0);
  attn_kernel<<<dim3(BS * NOBJ, NHEAD), 256, ATTN_SMEM, stream>>>(
      Qo, Km, Vm, O1, NOBJ, 0.125f);
  __bf16* X1 = (__bf16*)A1;  // Xm dead
  gemm(O1, blk[0].wo, X1, NT_O, 512, 512, nullptr, Xo, 0);
  __bf16* X1n = (__bf16*)A2;  // Km dead
  ln_kernel<<<NT_O, 256, 0, stream>>>(X1, blk[0].ln1g, blk[0].ln1b, X1n, 0);
  __bf16* Hf = (__bf16*)A6;
  gemm(X1n, blk[0].w1, Hf, NT_O, DINNER, 512, blk[0].b1, nullptr, 1);
  __bf16* X2 = (__bf16*)A3;  // Vm dead
  gemm(Hf, blk[0].w2, X2, NT_O, 512, DINNER, blk[0].b2, X1n, 0);
  __bf16* Mb = (__bf16*)A4;  // Qo dead
  ln_kernel<<<NT_O, 256, 0, stream>>>(X2, blk[0].ln2g, blk[0].ln2b, Mb, 0);

  // ================= Block 2 (p_inter) =================
  __bf16* Xi = (__bf16*)A0;  // Xo dead
  embed_kernel<<<NT_I, 256, 0, stream>>>(interaction_ids, emb, pos, Xi, LINT);
  __bf16* Mc = (__bf16*)A1;  // X1 dead
  mcat_kernel<<<NT_MC, 256, 0, stream>>>(Mb, Mc);
  __bf16* Kc = (__bf16*)A3;  // X2 dead
  __bf16* Vc = (__bf16*)A5;  // O1 dead
  __bf16* Qi = (__bf16*)A2;  // X1n dead
  gemm(Mc, blk[1].wk, Kc, NT_MC, 512, 512, nullptr, nullptr, 0);
  gemm(Mc, blk[1].wv, Vc, NT_MC, 512, 512, nullptr, nullptr, 0);
  gemm(Xi, blk[1].wq, Qi, NT_I, 512, 512, nullptr, nullptr, 0);
  __bf16* O2 = (__bf16*)A6;  // Hf dead
  attn_kernel<<<dim3(BS * (NOBJ - 1), NHEAD), 256, ATTN_SMEM, stream>>>(
      Qi, Kc, Vc, O2, 1, 0.125f);
  __bf16* X1b = (__bf16*)A4;  // Mb dead (Mcat built)
  gemm(O2, blk[1].wo, X1b, NT_I, 512, 512, nullptr, Xi, 0);
  __bf16* X1nb = (__bf16*)A2;  // Qi dead
  ln_kernel<<<NT_I, 256, 0, stream>>>(X1b, blk[1].ln1g, blk[1].ln1b, X1nb, 0);
  __bf16* Hfb = (__bf16*)A6;  // O2 dead
  gemm(X1nb, blk[1].w1, Hfb, NT_I, DINNER, 512, blk[1].b1, nullptr, 1);
  __bf16* X2b = (__bf16*)A3;  // Kc dead
  gemm(Hfb, blk[1].w2, X2b, NT_I, 512, DINNER, blk[1].b2, X1nb, 0);
  __bf16* Ibuf = (__bf16*)A5;  // Vc dead
  ln_kernel<<<NT_I, 256, 0, stream>>>(X2b, blk[1].ln2g, blk[1].ln2b, Ibuf, 0);

  // ========== Block 3 (p_road): KV length 1 => attention collapses =========
  __bf16* Xmap = (__bf16*)A0;  // Xi dead
  __bf16* vh = (__bf16*)(A0 + 65536);
  __bf16* rrow = (__bf16*)(A0 + 131072);
  f32_to_bf16_kernel<<<32, 256, 0, stream>>>(map_embeds, Xmap, BS * 512);
  gemm(Xmap, blk[2].wv, vh, BS, 512, 512, nullptr, nullptr, 0);
  gemm(vh, blk[2].wo, rrow, BS, 512, 512, nullptr, nullptr, 0);
  __bf16* X3 = (__bf16*)A4;  // X1b dead
  bcast_add_kernel<<<NT_I, 256, 0, stream>>>(Ibuf, rrow, X3);
  __bf16* X3n = (__bf16*)A2;  // X1nb dead
  ln_kernel<<<NT_I, 256, 0, stream>>>(X3, blk[2].ln1g, blk[2].ln1b, X3n, 0);
  __bf16* Hf3 = (__bf16*)A6;
  gemm(X3n, blk[2].w1, Hf3, NT_I, DINNER, 512, blk[2].b1, nullptr, 1);
  __bf16* X4 = (__bf16*)A3;
  gemm(Hf3, blk[2].w2, X4, NT_I, 512, DINNER, blk[2].b2, X3n, 0);
  // final LN writes f32 straight to d_out (16, 1984, 512)
  ln_kernel<<<NT_I, 256, 0, stream>>>(X4, blk[2].ln2g, blk[2].ln2b, d_out, 1);
}
